// KFGN_Mamba_74706661146793
// MI455X (gfx1250) — matrix operand order repaired and checked by reading, not compiled, not run on hardware
//
#include <hip/hip_runtime.h>
#include <hip/hip_bf16.h>

typedef __attribute__((ext_vector_type(16))) _Float16 v16h;
typedef __attribute__((ext_vector_type(8)))  _Float16 v8h;
typedef __attribute__((ext_vector_type(8)))  float    v8f;
typedef _Float16 h16;

// ---------------------------------------------------------------------------
// Model dims
// ---------------------------------------------------------------------------
#define BB      8
#define LL      512
#define FF      256
#define ROWS    (BB*LL)          // 4096
#define DIN     512              // D_INNER
#define DSTATE  16
#define DTRANK  16

// ---------------------------------------------------------------------------
// Small helpers
// ---------------------------------------------------------------------------
__device__ __forceinline__ float sigm_(float x) { return 1.f / (1.f + expf(-x)); }

// A fragment loader: halves 0..15 = K[kb..kb+15]
__device__ __forceinline__ v16h ldA_(const h16* p) {
  v8h lo = *(const v8h*)p;
  v8h hi = *(const v8h*)(p + 8);
  return __builtin_shufflevector(lo, hi, 0, 1, 2, 3, 4, 5, 6, 7,
                                         8, 9, 10, 11, 12, 13, 14, 15);
}

__device__ __forceinline__ v8f wmma_(v16h a, v16h b, v8f c) {
  return __builtin_amdgcn_wmma_f32_16x16x32_f16(false, a, false, b, (short)0, c, false, false);
}

__device__ __forceinline__ void store_tile_(v8f acc, float* __restrict__ C,
                                            const float* __restrict__ bias,
                                            const float* __restrict__ addend,
                                            int act, int rbase, int col, int N) {
  float bv = bias ? bias[col] : 0.f;
#pragma unroll
  for (int v = 0; v < 8; ++v) {
    float val = acc[v] + bv;
    size_t off = (size_t)(rbase + v) * N + col;
    if (addend) val += addend[off];
    if (act == 1)      val = sigm_(val);
    else if (act == 2) val = tanhf(val);
    else if (act == 3) val = val * sigm_(val);
    C[off] = val;
  }
}

// ---------------------------------------------------------------------------
// Adjacency preprocessing: colsum -> normA -> clipped matrix powers (fp32 exact)
// ---------------------------------------------------------------------------
__global__ void k_colinv(const float* __restrict__ A, float* __restrict__ colinv) {
  int i = threadIdx.x;
  float s = 0.f;
  for (int r = 0; r < 256; ++r) s += A[r * 256 + i];
  colinv[i] = 1.f / s;
}

__global__ void k_scale_rows(const float* __restrict__ A, const float* __restrict__ colinv,
                             float* __restrict__ normA) {
  int idx = blockIdx.x * 256 + threadIdx.x;
  normA[idx] = A[idx] * colinv[idx >> 8];
}

__global__ void k_min1(const float* __restrict__ src, float* __restrict__ dst) {
  int idx = blockIdx.x * 256 + threadIdx.x;
  dst[idx] = fminf(src[idx], 1.f);
}

__global__ void k_gemm256_min1(const float* __restrict__ A, const float* __restrict__ Bm,
                               float* __restrict__ C) {
  int idx = blockIdx.x * 256 + threadIdx.x;
  int i = idx >> 8, j = idx & 255;
  float acc = 0.f;
  for (int k = 0; k < 256; ++k) acc += A[i * 256 + k] * Bm[k * 256 + j];
  C[idx] = fminf(acc, 1.f);
}

// ---------------------------------------------------------------------------
// Conversions
// ---------------------------------------------------------------------------
__global__ void k_f32_to_h(const float* __restrict__ x, h16* __restrict__ y, int n) {
  int i = blockIdx.x * 256 + threadIdx.x;
  if (i < n) y[i] = (h16)x[i];
}

// W [K,N] fp32 -> Bt [N,K] f16  (WMMA B operand source; contiguous per-lane reads)
__global__ void k_transpose_h(const float* __restrict__ W, h16* __restrict__ Bt, int K, int N) {
  int idx = blockIdx.x * 256 + threadIdx.x;
  if (idx >= K * N) return;
  int k = idx / N, n = idx - k * N;
  Bt[(size_t)n * K + k] = (h16)W[idx];
}

// ---------------------------------------------------------------------------
// WMMA GEMM: C[M,N] = epilogue(Ah[M,K] @ Bt[N,K]^T + bias [+ addend])
// One 32x32 output macro-tile per wave32 (2x2 register blocking, 4 accumulators):
// 2 A-frags + 2 B-frags feed 4 WMMAs -> 32 B/lane/WMMA operand traffic.
// Wave-uniform half-width path for N % 32 == 16 (keeps EXEC all-ones for WMMA).
// K multiple of 32, M multiple of 32, N multiple of 16.
// act: 0=none 1=sigmoid 2=tanh 3=silu
// ---------------------------------------------------------------------------
__global__ void k_wmma_gemm(const h16* __restrict__ Ah, const h16* __restrict__ Bth,
                            const float* __restrict__ bias, const float* __restrict__ addend,
                            float* __restrict__ C, int M, int N, int K, int act,
                            int tilesN32, int waves) {
  int wave = (blockIdx.x << 3) + (threadIdx.x >> 5);
  if (wave >= waves) return;                    // uniform per wave: EXEC stays all-ones
  int lane = threadIdx.x & 31;
  int tm = wave / tilesN32;
  int tn = wave - tm * tilesN32;
  int m0 = tm << 5, n0 = tn << 5;
  bool hasN1 = (n0 + 16) < N;                   // wave-uniform
  int hi = lane >> 4, l15 = lane & 15;

  const h16* arow0 = Ah + (size_t)(m0 + l15) * K + (hi << 4);   // rows m0..m0+15
  const h16* arow1 = arow0 + (size_t)16 * K;                    // rows m0+16..m0+31
  const h16* brow0 = Bth + (size_t)(n0 + l15) * K + (hi << 4);  // cols n0..n0+15
  const h16* brow1 = brow0 + (size_t)16 * K;                    // cols n0+16..n0+31

  v8f acc00 = {}, acc01 = {}, acc10 = {}, acc11 = {};
  if (hasN1) {
    for (int k = 0; k < K; k += 32) {
      __builtin_prefetch(arow0 + k + 64, 0, 3);
      __builtin_prefetch(arow1 + k + 64, 0, 3);
      v16h a0 = ldA_(arow0 + k);
      v16h a1 = ldA_(arow1 + k);
      v16h b0 = *(const v16h*)(brow0 + k);
      v16h b1 = *(const v16h*)(brow1 + k);
      acc00 = wmma_(a0, b0, acc00);
      acc10 = wmma_(a1, b0, acc10);
      acc01 = wmma_(a0, b1, acc01);
      acc11 = wmma_(a1, b1, acc11);
    }
  } else {
    for (int k = 0; k < K; k += 32) {
      __builtin_prefetch(arow0 + k + 64, 0, 3);
      __builtin_prefetch(arow1 + k + 64, 0, 3);
      v16h a0 = ldA_(arow0 + k);
      v16h a1 = ldA_(arow1 + k);
      v16h b0 = *(const v16h*)(brow0 + k);
      acc00 = wmma_(a0, b0, acc00);
      acc10 = wmma_(a1, b0, acc10);
    }
  }

  int col = n0 + l15;
  int rbase = m0 + (hi << 3);                   // C: VGPR v -> row rbase + v
  store_tile_(acc00, C, bias, addend, act, rbase,      col,      N);
  store_tile_(acc10, C, bias, addend, act, rbase + 16, col,      N);
  if (hasN1) {
    store_tile_(acc01, C, bias, addend, act, rbase,      col + 16, N);
    store_tile_(acc11, C, bias, addend, act, rbase + 16, col + 16, N);
  }
}

// ---------------------------------------------------------------------------
// RMSNorm over last dim (256), fp32 out + f16 copy (next GEMM's A operand)
// ---------------------------------------------------------------------------
__global__ void k_rmsnorm(const float* __restrict__ x, const float* __restrict__ w,
                          float* __restrict__ y, h16* __restrict__ yh) {
  __shared__ float s[256];
  int row = blockIdx.x, t = threadIdx.x;
  float v = x[row * 256 + t];
  s[t] = v * v;
  __syncthreads();
  for (int st = 128; st > 0; st >>= 1) {
    if (t < st) s[t] += s[t + st];
    __syncthreads();
  }
  float r = rsqrtf(s[0] * (1.f / 256.f) + 1e-5f);
  float o = v * r * w[t];
  y[row * 256 + t] = o;
  yh[row * 256 + t] = (h16)o;
}

// ---------------------------------------------------------------------------
// adj_i = A_list[i] * sigmoid(gc_filters[i]); write transposed f16 (B operand)
// and keep adjs[2] fp32 for the selective-scan pad matrix.
// ---------------------------------------------------------------------------
__global__ void k_adj_prep(const float* __restrict__ Alist, const float* __restrict__ filt,
                           h16* __restrict__ adjT, float* __restrict__ adj3) {
  int idx = blockIdx.x * 256 + threadIdx.x;     // 3*65536
  int i = idx >> 16, rc = idx & 65535;
  int k = rc >> 8, n = rc & 255;
  float adj = Alist[idx] * sigm_(filt[idx]);
  adjT[(i << 16) + (n << 8) + k] = (h16)adj;
  if (i == 2) adj3[rc] = adj;
}

// ---------------------------------------------------------------------------
// combined = cat([gc1;gc2;gc3;0], dim=1).reshape(b,t,4f)  -> f16 [4096,1024]
// combined[b,t,j] pulls row r=4t+(j/256) of the 4t-row concatenation.
// ---------------------------------------------------------------------------
__global__ void k_combined(const float* __restrict__ gc, h16* __restrict__ out) {
  int idx = blockIdx.x * 256 + threadIdx.x;     // 4096*1024
  int row = idx >> 10, j = idx & 1023;
  int b = row >> 9, t = row & 511;
  int q = j >> 8, col = j & 255;
  int r = 4 * t + q;
  float v = 0.f;
  if (r < 1536) {
    int which = r >> 9, trow = r & 511;
    v = gc[(size_t)which * (ROWS * FF) + (size_t)(b * 512 + trow) * 256 + col];
  }
  out[idx] = (h16)v;
}

// ---------------------------------------------------------------------------
// Deterministic two-stage variance (ddof=1) reductions
// ---------------------------------------------------------------------------
__global__ void k_var_partial(const float* __restrict__ x, int n, float* __restrict__ part) {
  __shared__ float ss[256], sq[256];
  int t = threadIdx.x;
  float s = 0.f, q = 0.f;
  for (int i = blockIdx.x * 256 + t; i < n; i += gridDim.x * 256) {
    float v = x[i];
    s += v; q += v * v;
  }
  ss[t] = s; sq[t] = q;
  __syncthreads();
  for (int st = 128; st > 0; st >>= 1) {
    if (t < st) { ss[t] += ss[t + st]; sq[t] += sq[t + st]; }
    __syncthreads();
  }
  if (t == 0) { part[2 * blockIdx.x] = ss[0]; part[2 * blockIdx.x + 1] = sq[0]; }
}

__global__ void k_var_final(const float* __restrict__ p1, int nb1, float n1,
                            const float* __restrict__ p2, int nb2, float n2,
                            float* __restrict__ varbuf) {
  __shared__ float ss[256], sq[256];
  int t = threadIdx.x;
  float s = 0.f, q = 0.f;
  for (int i = t; i < nb1; i += 256) { s += p1[2 * i]; q += p1[2 * i + 1]; }
  ss[t] = s; sq[t] = q;
  __syncthreads();
  for (int st = 128; st > 0; st >>= 1) {
    if (t < st) { ss[t] += ss[t + st]; sq[t] += sq[t + st]; }
    __syncthreads();
  }
  if (t == 0) { float mean = ss[0] / n1; varbuf[0] = (sq[0] - mean * ss[0]) / (n1 - 1.f); }
  __syncthreads();
  s = 0.f; q = 0.f;
  for (int i = t; i < nb2; i += 256) { s += p2[2 * i]; q += p2[2 * i + 1]; }
  ss[t] = s; sq[t] = q;
  __syncthreads();
  for (int st = 128; st > 0; st >>= 1) {
    if (t < st) { ss[t] += ss[t + st]; sq[t] += sq[t + st]; }
    __syncthreads();
  }
  if (t == 0) { float mean = ss[0] / n2; varbuf[1] = (sq[0] - mean * ss[0]) / (n2 - 1.f); }
}

// ---------------------------------------------------------------------------
// pred = (og*tanh(ig*Cg)*var1*c + sig_out*var2) / (var1 + var2*c)
// (fg branch is dead: cell0 == 0 -> NC == 0)
// ---------------------------------------------------------------------------
__global__ void k_pred(const float* __restrict__ ig, const float* __restrict__ og,
                       const float* __restrict__ Cg, const float* __restrict__ sg,
                       const float* __restrict__ varbuf, const float* __restrict__ cptr,
                       float* __restrict__ x2, h16* __restrict__ x2h) {
  int i = blockIdx.x * 256 + threadIdx.x;
  float var1 = varbuf[0], var2 = varbuf[1], c = cptr[0];
  float hidden = og[i] * tanhf(ig[i] * Cg[i]);
  float o = (hidden * var1 * c + sg[i] * var2) / (var1 + var2 * c);
  x2[i] = o;
  x2h[i] = (h16)o;
}

// ---------------------------------------------------------------------------
// Causal depthwise conv (k=4, left pad) + SiLU. xmres holds [xm | res] (1024 cols).
// ---------------------------------------------------------------------------
__global__ void k_dwconv(const float* __restrict__ xmres, const float* __restrict__ w,
                         const float* __restrict__ bconv, float* __restrict__ xmc,
                         h16* __restrict__ xmch) {
  int idx = blockIdx.x * 256 + threadIdx.x;     // 4096*512
  int row = idx >> 9, d = idx & 511;
  int b = row >> 9, l = row & 511;
  float acc = bconv[d];
#pragma unroll
  for (int i = 0; i < 4; ++i) {
    int ll = l - 3 + i;
    if (ll >= 0) acc += xmres[(size_t)(b * 512 + ll) * 1024 + d] * w[d * 4 + i];
  }
  float o = acc * sigm_(acc);
  xmc[idx] = o;
  xmch[idx] = (h16)o;
}

// ---------------------------------------------------------------------------
// delta = softplus(x_dbl[:, :16] @ dt_w + dt_b)  (K=16 < WMMA depth: plain FMA)
// ---------------------------------------------------------------------------
__global__ void k_delta(const float* __restrict__ x_dbl, const float* __restrict__ dt_w,
                        const float* __restrict__ dt_b, h16* __restrict__ delta_h) {
  int idx = blockIdx.x * 256 + threadIdx.x;     // 4096*512
  int row = idx >> 9, n = idx & 511;
  const float* xr = x_dbl + (size_t)row * 48;
  float acc = dt_b[n];
#pragma unroll
  for (int r = 0; r < 16; ++r) acc += xr[r] * dt_w[r * 512 + n];
  acc = (acc > 20.f) ? acc : log1pf(expf(acc));
  delta_h[idx] = (h16)acc;
}

// pad matrix (ones, top-left 256x256 = temp_adj), transposed f16 [512,512]
__global__ void k_padT(const float* __restrict__ adj3, h16* __restrict__ padT) {
  int idx = blockIdx.x * 256 + threadIdx.x;     // 512*512
  int n = idx >> 9, k = idx & 511;
  float v = (n < 256 && k < 256) ? adj3[k * 256 + n] : 1.f;
  padT[idx] = (h16)v;
}

// ---------------------------------------------------------------------------
// Selective scan: one thread per (b, d_inner); 512 sequential steps, 16 states.
// ---------------------------------------------------------------------------
__global__ void k_scan(const float* __restrict__ delta_p, const float* __restrict__ u,
                       const float* __restrict__ x_dbl, const float* __restrict__ A_log,
                       const float* __restrict__ Dp, float* __restrict__ y) {
  int tid = blockIdx.x * 256 + threadIdx.x;     // 4096
  int b = tid >> 9, d = tid & 511;
  float Arow[DSTATE];
#pragma unroll
  for (int n = 0; n < DSTATE; ++n) Arow[n] = -expf(A_log[d * DSTATE + n]);
  float st[DSTATE];
#pragma unroll
  for (int n = 0; n < DSTATE; ++n) st[n] = 0.f;
  float Dpd = Dp[d];
  for (int l = 0; l < LL; ++l) {
    size_t row = (size_t)b * LL + l;
    float dp = delta_p[row * DIN + d];
    float uu = u[row * DIN + d];
    const float* BC = x_dbl + row * 48 + DTRANK;
    float acc = 0.f;
#pragma unroll
    for (int n = 0; n < DSTATE; ++n) {
      float dA = expf(dp * Arow[n]);
      st[n] = dA * st[n] + dp * BC[n] * uu;
      acc += st[n] * BC[DSTATE + n];
    }
    y[row * DIN + d] = acc + uu * Dpd;
  }
}

// yg = y * silu(res) as f16 (A operand of out_proj GEMM)
__global__ void k_gate(const float* __restrict__ y, const float* __restrict__ xmres,
                       h16* __restrict__ ygh) {
  int idx = blockIdx.x * 256 + threadIdx.x;     // 4096*512
  int row = idx >> 9, d = idx & 511;
  float res = xmres[(size_t)row * 1024 + 512 + d];
  ygh[idx] = (h16)(y[idx] * (res * sigm_(res)));
}

// ---------------------------------------------------------------------------
// Host-side orchestration
// ---------------------------------------------------------------------------
extern "C" void kernel_launch(void* const* d_in, const int* in_sizes, int n_in,
                              void* d_out, int out_size, void* d_ws, size_t ws_size,
                              hipStream_t stream) {
  (void)in_sizes; (void)n_in; (void)out_size;

  const float* input_ids    = (const float*)d_in[0];
  const float* A_adj        = (const float*)d_in[1];
  // params pytree (dict keys sorted): dec_b, dec_w, enc_b, enc_w, final_norm_w, layers[2]
  const float* dec_b        = (const float*)d_in[2];
  const float* dec_w        = (const float*)d_in[3];
  const float* enc_b        = (const float*)d_in[4];
  const float* enc_w        = (const float*)d_in[5];
  const float* final_norm_w = (const float*)d_in[6];
  // layer leaves sorted: A_log,Cl_b,Cl_w,Dp,c,conv_b,conv_w,dt_b,dt_w,fl_b,fl_w,
  //                      gc_filters,il_b,il_w,in_proj,nbr_w,norm_w,ol_b,ol_w,
  //                      out_proj,sig_b,sig_w,x_proj
  auto LPF = [&](int layer, int leaf) -> const float* {
    return (const float*)d_in[7 + layer * 23 + leaf];
  };

  // ---- workspace layout (bump allocator, 256B aligned) ----
  char* base = (char*)d_ws;
  size_t off = 0;
  auto alloc = [&](size_t bytes) -> char* {
    char* p = base + off;
    off += (bytes + 255) & ~(size_t)255;
    return p;
  };
  const size_t GCSZ = (size_t)ROWS * FF;              // 1,048,576 elems

  float* colinv  = (float*)alloc(256 * 4);
  float* normA   = (float*)alloc(65536 * 4);
  float* Alist   = (float*)alloc(3 * 65536 * 4);      // A1,A2,A3 contiguous
  float* adj3    = (float*)alloc(65536 * 4);
  h16*   adjT    = (h16*)  alloc(3 * 65536 * 2);
  float* x       = (float*)alloc(GCSZ * 4);           // residual stream
  float* x1      = (float*)alloc(GCSZ * 4);
  h16*   x1h     = (h16*)  alloc(GCSZ * 2);
  // R1: gc1..gc3 (phase A) -> ig/og/Cg + sig (phase B) -> xmres [4096,1024] (phase C)
  float* R1      = (float*)alloc(4 * GCSZ * 4);
  // R2: combined f16 [4096,1024] (phase A) -> y f32 [4096,512] (phase B)
  char*  R2      = alloc((size_t)ROWS * 1024 * 2);
  float* x2      = (float*)alloc(GCSZ * 4);
  h16*   x2h     = (h16*)  alloc(GCSZ * 2);
  h16*   Bt      = (h16*)  alloc(262144 * 2);         // largest weight^T / padT
  float* xmc     = (float*)alloc((size_t)ROWS * DIN * 4);
  h16*   xmch    = (h16*)  alloc((size_t)ROWS * DIN * 2);
  float* x_dbl   = (float*)alloc((size_t)ROWS * 48 * 4);
  h16*   delta_h = (h16*)  alloc((size_t)ROWS * DIN * 2);
  float* delta_p = (float*)alloc((size_t)ROWS * DIN * 4);
  h16*   ygh     = (h16*)  alloc((size_t)ROWS * DIN * 2);
  float* part1   = (float*)alloc(1024 * 2 * 4);
  float* part2   = (float*)alloc(1024 * 2 * 4);
  float* varbuf  = (float*)alloc(64);
  if (off > ws_size) return;                          // not enough scratch

  float* gc     = R1;                                 // 3 * GCSZ
  float* igb    = R1;
  float* ogb    = R1 + GCSZ;
  float* Cgb    = R1 + 2 * GCSZ;
  float* sigb   = R1 + 3 * GCSZ;
  float* xmres  = R1;                                 // [4096,1024]
  h16*   combined = (h16*)R2;
  float* ybuf     = (float*)R2;

  auto gemm = [&](const h16* Ah, const h16* Bth, const float* bias, const float* addend,
                  float* C, int M, int N, int K, int act) {
    int tilesN32 = (N + 31) / 32;
    int waves = (M / 32) * tilesN32;
    int blocks = (waves + 7) / 8;                     // 8 waves / 256-thread block
    k_wmma_gemm<<<blocks, 256, 0, stream>>>(Ah, Bth, bias, addend, C, M, N, K, act,
                                            tilesN32, waves);
  };
  auto transpose = [&](const float* W, h16* Bth, int K, int N) {
    int n = K * N;
    k_transpose_h<<<(n + 255) / 256, 256, 0, stream>>>(W, Bth, K, N);
  };

  // ---- adjacency powers ----
  k_colinv<<<1, 256, 0, stream>>>(A_adj, colinv);
  k_scale_rows<<<256, 256, 0, stream>>>(A_adj, colinv, normA);
  k_min1<<<256, 256, 0, stream>>>(normA, Alist);                                 // A1
  k_gemm256_min1<<<256, 256, 0, stream>>>(Alist, normA, Alist + 65536);          // A2
  k_gemm256_min1<<<256, 256, 0, stream>>>(Alist + 65536, normA, Alist + 131072); // A3

  // ---- encoder: x = input_ids @ enc_w + enc_b ----
  k_f32_to_h<<<(int)(GCSZ / 256), 256, 0, stream>>>(input_ids, x1h, (int)GCSZ);
  transpose(enc_w, Bt, 256, 256);
  gemm(x1h, Bt, enc_b, nullptr, x, ROWS, 256, 256, 0);

  for (int li = 0; li < 2; ++li) {
    const float* A_log    = LPF(li, 0);
    const float* Cl_b     = LPF(li, 1);
    const float* Cl_w     = LPF(li, 2);
    const float* Dpv      = LPF(li, 3);
    const float* c_scalar = LPF(li, 4);
    const float* conv_b   = LPF(li, 5);
    const float* conv_w   = LPF(li, 6);
    const float* dt_b     = LPF(li, 7);
    const float* dt_w     = LPF(li, 8);
    const float* gc_filt  = LPF(li, 11);
    const float* il_b     = LPF(li, 12);
    const float* il_w     = LPF(li, 13);
    const float* in_proj  = LPF(li, 14);
    const float* norm_w   = LPF(li, 16);
    const float* ol_b     = LPF(li, 17);
    const float* ol_w     = LPF(li, 18);
    const float* out_proj = LPF(li, 19);
    const float* sig_b    = LPF(li, 20);
    const float* sig_w    = LPF(li, 21);
    const float* x_proj   = LPF(li, 22);

    // x1 = rmsnorm(x)
    k_rmsnorm<<<ROWS, 256, 0, stream>>>(x, norm_w, x1, x1h);

    // adjs + gc GEMMs
    k_adj_prep<<<768, 256, 0, stream>>>(Alist, gc_filt, adjT, adj3);
    for (int i = 0; i < 3; ++i)
      gemm(x1h, adjT + i * 65536, nullptr, nullptr, gc + i * GCSZ, ROWS, 256, 256, 0);

    // variances (before gc region is recycled)
    k_var_partial<<<1024, 256, 0, stream>>>(x1, (int)GCSZ, part1);
    k_var_partial<<<1024, 256, 0, stream>>>(gc, (int)(3 * GCSZ), part2);
    k_var_final<<<1, 256, 0, stream>>>(part1, 1024, (float)GCSZ,
                                       part2, 1024, (float)(3 * GCSZ), varbuf);

    // combined f16 [4096,1024]
    k_combined<<<16384, 256, 0, stream>>>(gc, combined);

    // sig_out = silu(x1 @ sig_w + sig_b)
    transpose(sig_w, Bt, 256, 256);
    gemm(x1h, Bt, sig_b, nullptr, sigb, ROWS, 256, 256, 3);

    // gates (fg is dead: cell0 == 0)
    transpose(il_w, Bt, 1024, 256);
    gemm(combined, Bt, il_b, nullptr, igb, ROWS, 256, 1024, 1);
    transpose(ol_w, Bt, 1024, 256);
    gemm(combined, Bt, ol_b, nullptr, ogb, ROWS, 256, 1024, 1);
    transpose(Cl_w, Bt, 1024, 256);
    gemm(combined, Bt, Cl_b, nullptr, Cgb, ROWS, 256, 1024, 2);

    // x2 = pred
    k_pred<<<(int)(GCSZ / 256), 256, 0, stream>>>(igb, ogb, Cgb, sigb, varbuf, c_scalar, x2, x2h);

    // mamba: in_proj -> [xm|res]
    transpose(in_proj, Bt, 256, 1024);
    gemm(x2h, Bt, nullptr, nullptr, xmres, ROWS, 1024, 256, 0);

    // causal dwconv + silu
    k_dwconv<<<8192, 256, 0, stream>>>(xmres, conv_w, conv_b, xmc, xmch);

    // x_dbl = xm @ x_proj  [4096,48]
    transpose(x_proj, Bt, 512, 48);
    gemm(xmch, Bt, nullptr, nullptr, x_dbl, ROWS, 48, 512, 0);

    // delta = softplus(x_dbl[:, :16] @ dt_w + dt_b) -> f16
    k_delta<<<8192, 256, 0, stream>>>(x_dbl, dt_w, dt_b, delta_h);

    // delta_p = delta @ pad  (pad = ones w/ temp_adj block)
    k_padT<<<1024, 256, 0, stream>>>(adj3, Bt);
    gemm(delta_h, Bt, nullptr, nullptr, delta_p, ROWS, 512, 512, 0);

    // selective scan
    k_scan<<<16, 256, 0, stream>>>(delta_p, xmc, x_dbl, A_log, Dpv, ybuf);

    // gate by silu(res), out_proj, add residual x1 -> new x
    k_gate<<<8192, 256, 0, stream>>>(ybuf, xmres, ygh);
    transpose(out_proj, Bt, 512, 256);
    gemm(ygh, Bt, nullptr, x1, x, ROWS, 256, 512, 0);
  }

  // ---- final norm + decoder ----
  k_rmsnorm<<<ROWS, 256, 0, stream>>>(x, final_norm_w, x1, x1h);
  transpose(dec_w, Bt, 256, 256);
  gemm(x1h, Bt, dec_b, nullptr, (float*)d_out, ROWS, 256, 256, 0);
}
